// ISTFT_83734682402944
// MI455X (gfx1250) — matrix operand order repaired and checked by reading, not compiled, not run on hardware
//
#include <hip/hip_runtime.h>
#include <hip/hip_bf16.h>

typedef _Float16 h16;
typedef __attribute__((ext_vector_type(16))) _Float16 v16h;
typedef __attribute__((ext_vector_type(8)))  _Float16 v8h;
typedef __attribute__((ext_vector_type(8)))  float    v8f;

#define BB      16
#define NFREQ   513
#define TT      2048
#define NFFT    1024
#define KPAD    1056      // 33 * 32
#define HOPSZ   256
#define PADSZ   384
#define OUTLEN  524288    // per-batch output samples (2^19)
#define SROW    40        // LDS row stride in halves (pad 32 -> 40 to avoid bank conflicts)

// ---------------------------------------------------------------------------
// Kernel 0: build split-f16 irfft weight matrix W[1024][1056].
//   cols 0..512   : c_k * cos(2*pi*k*n/1024), c_0=c_512=1 else 2
//   cols 513..1025: -2 * sin(2*pi*k*n/1024)  (k = col-513; exactly 0 at k=0,512)
//   cols 1026..   : zero padding
// Exact angle reduction: (k*n) mod 1024, then cospif/sinpif.
// ---------------------------------------------------------------------------
__global__ void build_w(h16* __restrict__ whi, h16* __restrict__ wlo) {
  int idx = blockIdx.x * blockDim.x + threadIdx.x;
  if (idx >= NFFT * KPAD) return;
  int n  = idx / KPAD;
  int kk = idx - n * KPAD;
  float w = 0.0f;
  if (kk < 513) {
    float c = (kk == 0 || kk == 512) ? 1.0f : 2.0f;
    int m = (kk * n) & 1023;
    w = c * cospif((float)m * (1.0f / 512.0f));
  } else if (kk < 1026) {
    int k = kk - 513;
    int m = (k * n) & 1023;
    w = -2.0f * sinpif((float)m * (1.0f / 512.0f));
  }
  h16 hi = (h16)w;
  float lo = w - (float)hi;
  whi[idx] = hi;
  wlo[idx] = (h16)lo;
}

__device__ __forceinline__ v16h pack16(v8h a, v8h b) {
  v16h r;
#pragma unroll
  for (int i = 0; i < 8; ++i) { r[i] = a[i]; r[i + 8] = b[i]; }
  return r;
}

// ---------------------------------------------------------------------------
// Kernel 1: frames[b][t][n] = (1/1024) * hann(n) * mask[b,t] * (W @ S)[n,t]
// GEMM M=1024, N=2048 (per b), K=1056; split-f16 (3 WMMAs per tile per k-step).
// WG = 256 thr = 8 waves arranged 4(M) x 2(N); wave tile 64x32 (4x2 C tiles).
// WG tile = 256(M) x 64(N)  -> each S tile is staged by only 4 workgroups.
// ---------------------------------------------------------------------------
__global__ void __launch_bounds__(256) istft_gemm(
    const float* __restrict__ spec_re, const float* __restrict__ spec_im,
    const float* __restrict__ mask,    const h16*  __restrict__ whi,
    const h16*  __restrict__ wlo,      float*      __restrict__ frames)
{
  __shared__ __align__(16) h16 s_hi[64 * SROW];
  __shared__ __align__(16) h16 s_lo[64 * SROW];

  const int tid    = threadIdx.x;
  const int b      = blockIdx.z;
  const int tBase  = blockIdx.x * 64;    // N (t) tile base
  const int mBase  = blockIdx.y * 256;   // M (n) tile base
  const int wave   = tid >> 5;
  const int lane   = tid & 31;
  const int laneM  = lane & 15;
  const int laneHi = lane >> 4;
  const int waveM  = wave >> 1;          // 0..3  -> 64 rows each
  const int waveN  = wave & 1;           // 0..1  -> 32 cols each

  v8f acc[4][2] = {};

  // A rows for the four M subtiles (A lane M = lane%16)
  const int nrow0 = mBase + waveM * 64 + laneM;
  size_t wOff[4];
#pragma unroll
  for (int ms = 0; ms < 4; ++ms) wOff[ms] = (size_t)(nrow0 + 16 * ms) * KPAD;

  for (int ks = 0; ks < KPAD / 32; ++ks) {
    const int kBase = ks * 32;
    __syncthreads();
    // ---- stage S tile (32 k x 64 t), f32 -> f16 hi/lo, transposed [t][k] in LDS
#pragma unroll
    for (int i = 0; i < 8; ++i) {
      int e  = tid + i * 256;
      int kl = e >> 6;      // 0..31
      int tl = e & 63;      // 0..63
      int kk = kBase + kl;
      float v = 0.0f;
      if (kk < 513)
        v = spec_re[((size_t)b * NFREQ + kk) * TT + tBase + tl];
      else if (kk < 1026)
        v = spec_im[((size_t)b * NFREQ + (kk - 513)) * TT + tBase + tl];
      h16 hi = (h16)v;
      float lo = v - (float)hi;
      s_hi[tl * SROW + kl] = hi;
      s_lo[tl * SROW + kl] = (h16)lo;
    }
    __syncthreads();

    // ---- A fragments from global W (fragment layout: h<8 -> K=h, h>=8 -> K=h+8; +8 for hi lanes)
    v16h ahi[4], alo[4];
#pragma unroll
    for (int ms = 0; ms < 4; ++ms) {
      const h16* pa = whi + wOff[ms] + kBase + laneHi * 8;
      const h16* qa = wlo + wOff[ms] + kBase + laneHi * 8;
      ahi[ms] = pack16(*(const v8h*)pa, *(const v8h*)(pa + 16));
      alo[ms] = pack16(*(const v8h*)qa, *(const v8h*)(qa + 16));
    }

    // ---- B fragments from LDS (lane N = lane%16; h<8 -> K=h, h>=8 -> K=h+8; +8 for hi lanes)
    v16h bhi[2], blo[2];
#pragma unroll
    for (int ns = 0; ns < 2; ++ns) {
      int tl = waveN * 32 + ns * 16 + laneM;
      const h16* pb = &s_hi[tl * SROW + laneHi * 8];
      const h16* qb = &s_lo[tl * SROW + laneHi * 8];
      bhi[ns] = pack16(*(const v8h*)pb, *(const v8h*)(pb + 16));
      blo[ns] = pack16(*(const v8h*)qb, *(const v8h*)(qb + 16));
    }

    // ---- split-precision product: Whi*Shi + Whi*Slo + Wlo*Shi
#pragma unroll
    for (int ms = 0; ms < 4; ++ms)
#pragma unroll
      for (int ns = 0; ns < 2; ++ns) {
        acc[ms][ns] = __builtin_amdgcn_wmma_f32_16x16x32_f16(
            false, ahi[ms], false, bhi[ns], (short)0, acc[ms][ns], false, false);
        acc[ms][ns] = __builtin_amdgcn_wmma_f32_16x16x32_f16(
            false, ahi[ms], false, blo[ns], (short)0, acc[ms][ns], false, false);
        acc[ms][ns] = __builtin_amdgcn_wmma_f32_16x16x32_f16(
            false, alo[ms], false, bhi[ns], (short)0, acc[ms][ns], false, false);
      }
  }

  // ---- epilogue: scale by (1/1024)*hann(n)*mask, store frames[b][t][n]
#pragma unroll
  for (int ns = 0; ns < 2; ++ns) {
    int t = tBase + waveN * 32 + ns * 16 + laneM;   // C lane col = lane%16
    float scale = mask[b * TT + t] * (1.0f / 1024.0f);
    float* dst0 = frames + (((size_t)b * TT + t) << 10);
#pragma unroll
    for (int ms = 0; ms < 4; ++ms) {
      int nb = mBase + waveM * 64 + ms * 16 + laneHi * 8;  // C VGPR r -> m = r + 8*(lane/16)
      float vals[8];
#pragma unroll
      for (int r = 0; r < 8; ++r) {
        int n = nb + r;
        float w = 0.5f - 0.5f * cospif((float)n * (1.0f / 512.0f));
        vals[r] = acc[ms][ns][r] * scale * w;
      }
      *(float4*)(dst0 + nb)     = make_float4(vals[0], vals[1], vals[2], vals[3]);
      *(float4*)(dst0 + nb + 4) = make_float4(vals[4], vals[5], vals[6], vals[7]);
    }
  }
}

// ---------------------------------------------------------------------------
// Kernel 2: overlap-add gather (<=4 frames per sample), envelope division,
// effective mask. Deterministic (no atomics).
// ---------------------------------------------------------------------------
__global__ void istft_fold(const float* __restrict__ frames,
                           const float* __restrict__ mask,
                           float* __restrict__ out)
{
  int gid = blockIdx.x * 256 + threadIdx.x;   // 0 .. BB*OUTLEN-1
  int b = gid >> 19;
  int j = gid & (OUTLEN - 1);
  int p   = j + PADSZ;          // position in untrimmed fold output
  int phi = p >> 8;             // floor(p / HOP)
  int pm  = p & 255;
  float y = 0.0f, env = 0.0f, msum = 0.0f;
#pragma unroll
  for (int d = 0; d < 4; ++d) {
    int t = phi - d;
    if (t >= 0 && t < TT) {
      int n = pm + (d << 8);    // 0..1023 always
      y += frames[(((size_t)b * TT + t) << 10) + n];
      float w = 0.5f - 0.5f * cospif((float)n * (1.0f / 512.0f));
      env  += w * w;
      msum += mask[b * TT + t];
    }
  }
  out[gid] = y / env;
  out[BB * OUTLEN + gid] = (msum > 0.0f) ? 1.0f : 0.0f;
}

extern "C" void kernel_launch(void* const* d_in, const int* in_sizes, int n_in,
                              void* d_out, int out_size, void* d_ws, size_t ws_size,
                              hipStream_t stream) {
  const float* spec_re = (const float*)d_in[0];
  const float* spec_im = (const float*)d_in[1];
  const float* mask    = (const float*)d_in[2];
  float* out = (float*)d_out;

  char* ws = (char*)d_ws;
  const size_t wbytes = (size_t)NFFT * KPAD * sizeof(h16);  // 2,162,688
  h16*   whi    = (h16*)ws;
  h16*   wlo    = (h16*)(ws + wbytes);
  float* frames = (float*)(ws + 2 * wbytes);                // 128 MB

  build_w<<<(NFFT * KPAD + 255) / 256, 256, 0, stream>>>(whi, wlo);

  dim3 grid(TT / 64, NFFT / 256, BB);
  istft_gemm<<<grid, 256, 0, stream>>>(spec_re, spec_im, mask, whi, wlo, frames);

  istft_fold<<<(BB * OUTLEN) / 256, 256, 0, stream>>>(frames, mask, out);
}
